// JambaMambaMixer_10445360464006
// MI455X (gfx1250) — compile-verified, matrix operand order checked by reading, not run on hardware
//
#include <hip/hip_runtime.h>
#include <hip/hip_bf16.h>

// ---------------------------------------------------------------------------
// Jamba Mamba mixer for MI455X (gfx1250):
//   bf16 WMMA GEMMs with double-buffered GLOBAL_LOAD_ASYNC_TO_LDS staging,
//   fused conv/rmsnorm/selective-scan elementwise kernels.
// ---------------------------------------------------------------------------

#define LL 2048      // sequence length
#define HH 4096      // hidden size
#define DD 8192      // intermediate size
#define NN 16        // ssm state size
#define KC 4         // conv kernel size
#define RR 256       // time-step rank
#define RN 288       // R + 2N

typedef __attribute__((ext_vector_type(16))) __bf16 v16bf;
typedef __attribute__((ext_vector_type(8)))  __bf16 v8bf;
typedef __attribute__((ext_vector_type(4)))  __bf16 v4bf;
typedef __attribute__((ext_vector_type(8)))  float  v8f;

static __device__ __forceinline__ __bf16 to_bf16(float f) { return (__bf16)f; }

// async DMA: global -> LDS, 16B per lane, tracked by ASYNCcnt
static __device__ __forceinline__ void async_ld16(unsigned lds_off,
                                                  unsigned long long gaddr) {
  asm volatile("global_load_async_to_lds_b128 %0, %1, off"
               :: "v"(lds_off), "v"(gaddr) : "memory");
}
static __device__ __forceinline__ void async_ld16_o16(unsigned lds_off,
                                                      unsigned long long gaddr) {
  asm volatile("global_load_async_to_lds_b128 %0, %1, off offset:16"
               :: "v"(lds_off), "v"(gaddr) : "memory");
}

// ---------------------------------------------------------------------------
// fp32 -> bf16 conversion (weights + input), 4 elems/thread
// ---------------------------------------------------------------------------
__global__ __launch_bounds__(256)
void f32_to_bf16_kernel(const float* __restrict__ src, __bf16* __restrict__ dst,
                        size_t n) {
  size_t i = ((size_t)blockIdx.x * 256 + threadIdx.x) * 4;
  if (i + 3 < n) {
    float4 v = *(const float4*)(src + i);
    v4bf o;
    o[0] = to_bf16(v.x);
    o[1] = to_bf16(v.y);
    o[2] = to_bf16(v.z);
    o[3] = to_bf16(v.w);
    *(v4bf*)(dst + i) = o;
  }
}

// ---------------------------------------------------------------------------
// Tiled bf16 WMMA GEMM:  C[M,N] = A[M,K] @ B[N,K]^T   (f32 accumulate)
// Block tile 128x128, BK=32, 8 waves: wave (wm in 0..3, wn in 0..1)
// computes a 32x64 sub-tile via 2x4 fragments of v_wmma_f32_16x16x32_bf16.
// LDS is double-buffered; tiles arrive via GLOBAL_LOAD_ASYNC_TO_LDS_B128
// so the next tile's DMA overlaps the current tile's WMMAs.
// EPI==1: C = softplus(C + bias[n])   (dt_proj epilogue)
// ---------------------------------------------------------------------------
#define BM 128
#define BN 128
#define BK 32

template <int EPI>
__global__ __launch_bounds__(256)
void gemm_bf16_wmma(const __bf16* __restrict__ A, const __bf16* __restrict__ Bw,
                    const float* __restrict__ bias, float* __restrict__ C,
                    int M, int N, int K) {
  __shared__ __bf16 sA[2][BM][BK];
  __shared__ __bf16 sB[2][BN][BK];

  const int tid  = threadIdx.x;
  const int lane = tid & 31;
  const int wave = tid >> 5;
  const int wm   = wave & 3;   // 4 waves along M (4*32 = 128)
  const int wn   = wave >> 2;  // 2 waves along N (2*64 = 128)
  const int m0   = blockIdx.y * BM;
  const int n0   = blockIdx.x * BN;

  v8f acc[2][4];
  #pragma unroll
  for (int i = 0; i < 2; i++)
    #pragma unroll
    for (int j = 0; j < 4; j++)
      #pragma unroll
      for (int e = 0; e < 8; e++) acc[i][j][e] = 0.f;

  // ISA 16-bit fragment lane layout: row = lane&15, k-base = (lane>>4)*8,
  // halves at [kb, kb+8) and [kb+16, kb+24).
  const int frow = lane & 15;
  const int kb   = (lane >> 4) << 3;

  // staging assignment: 2 threads per row, 32 bytes (16 bf16) each
  const int sr = tid >> 1;
  const int sc = (tid & 1) << 4;
  // out-of-range B rows clamp to a valid row; the garbage values only reach
  // accumulator lanes whose output column fails the n<N store guard.
  const int nrow = n0 + sr;
  const int nclp = (nrow < N) ? nrow : (N - 1);

  const __bf16* aptr = A  + (size_t)(m0 + sr) * K + sc;
  const __bf16* bptr = Bw + (size_t)nclp     * K + sc;

  // generic pointers into LDS: low 32 bits are the LDS byte offset
  unsigned ldsA[2], ldsB[2];
  ldsA[0] = (unsigned)(uintptr_t)&sA[0][sr][sc];
  ldsA[1] = (unsigned)(uintptr_t)&sA[1][sr][sc];
  ldsB[0] = (unsigned)(uintptr_t)&sB[0][sr][sc];
  ldsB[1] = (unsigned)(uintptr_t)&sB[1][sr][sc];

  auto issue_tile = [&](int buf, int k0) {
    unsigned long long ga = (unsigned long long)(uintptr_t)(aptr + k0);
    unsigned long long gb = (unsigned long long)(uintptr_t)(bptr + k0);
    async_ld16(ldsA[buf], ga);
    async_ld16_o16(ldsA[buf], ga);
    async_ld16(ldsB[buf], gb);
    async_ld16_o16(ldsB[buf], gb);
  };

  const int nk = K / BK;
  issue_tile(0, 0);

  for (int ki = 0; ki < nk; ki++) {
    const int cur = ki & 1;
    if (ki + 1 < nk) {
      issue_tile(cur ^ 1, (ki + 1) * BK);
      // retire the 4 older (current-tile) transfers; next tile stays in flight
      asm volatile("s_wait_asynccnt 4" ::: "memory");
    } else {
      asm volatile("s_wait_asynccnt 0" ::: "memory");
    }
    __syncthreads();

    v16bf afrag[2];
    #pragma unroll
    for (int fm = 0; fm < 2; fm++) {
      union { v16bf v; v8bf h[2]; } ua;
      ua.h[0] = *(const v8bf*)(&sA[cur][wm * 32 + fm * 16 + frow][kb]);
      ua.h[1] = *(const v8bf*)(&sA[cur][wm * 32 + fm * 16 + frow][kb + 16]);
      afrag[fm] = ua.v;
    }
    #pragma unroll
    for (int fn = 0; fn < 4; fn++) {
      union { v16bf v; v8bf h[2]; } ub;
      ub.h[0] = *(const v8bf*)(&sB[cur][wn * 64 + fn * 16 + frow][kb]);
      ub.h[1] = *(const v8bf*)(&sB[cur][wn * 64 + fn * 16 + frow][kb + 16]);
      v16bf bfrag = ub.v;
      #pragma unroll
      for (int fm = 0; fm < 2; fm++) {
        acc[fm][fn] = __builtin_amdgcn_wmma_f32_16x16x32_bf16(
            false, afrag[fm], false, bfrag, (short)0, acc[fm][fn],
            false, false);
      }
    }
    __syncthreads();  // all waves done reading buf `cur` before it is refilled
  }

  // ---- epilogue + store (C layout: vgpr r, lanes 0-15 -> M=r, 16-31 -> M=r+8) ----
  const int moff = (lane >> 4) << 3;
  const int ncol = lane & 15;
  #pragma unroll
  for (int fm = 0; fm < 2; fm++) {
    #pragma unroll
    for (int fn = 0; fn < 4; fn++) {
      const int n = n0 + wn * 64 + fn * 16 + ncol;
      if (n < N) {
        #pragma unroll
        for (int r = 0; r < 8; r++) {
          const int m = m0 + wm * 32 + fm * 16 + moff + r;
          float v = acc[fm][fn][r];
          if (EPI == 1) {
            v += bias[n];
            v = (v > 20.f) ? v : log1pf(__expf(v));
          }
          C[(size_t)m * N + n] = v;
        }
      }
    }
  }
}

// ---------------------------------------------------------------------------
// Causal depthwise conv1d (K=4) + bias + SiLU.  proj row stride = 2*DD.
// Writes f32 (for scan / D-skip) and bf16 (for x_proj GEMM A matrix).
// ---------------------------------------------------------------------------
__global__ __launch_bounds__(256)
void conv_silu_kernel(const float* __restrict__ proj, const float* __restrict__ cw,
                      const float* __restrict__ cb, float* __restrict__ uf,
                      __bf16* __restrict__ ub) {
  const int idx = blockIdx.x * 256 + threadIdx.x;  // t*DD + d
  const int t = idx >> 13;
  const int d = idx & (DD - 1);
  float acc = cb[d];
  #pragma unroll
  for (int j = 0; j < KC; j++) {
    const int tt = t - (KC - 1) + j;
    if (tt >= 0) acc += proj[(size_t)tt * (2 * DD) + d] * cw[d * KC + j];
  }
  const float s = acc / (1.f + __expf(-acc));  // SiLU
  uf[idx] = s;
  ub[idx] = to_bf16(s);
}

// ---------------------------------------------------------------------------
// RMSNorms on dt (256), B (16), C (16).  One wave per sequence position.
// ---------------------------------------------------------------------------
__global__ __launch_bounds__(256)
void rmsnorm_kernel(const float* __restrict__ ssm, const float* __restrict__ dtw,
                    const float* __restrict__ bw, const float* __restrict__ cwn,
                    __bf16* __restrict__ dt_out, float* __restrict__ Bn,
                    float* __restrict__ Cn) {
  const int wave = threadIdx.x >> 5;
  const int lane = threadIdx.x & 31;
  const int t = blockIdx.x * 8 + wave;
  const float* row = ssm + (size_t)t * RN;

  // ---- dt: 256 elems, 8 per lane ----
  float x[8];
  float ss = 0.f;
  #pragma unroll
  for (int j = 0; j < 8; j++) {
    x[j] = row[lane + j * 32];
    ss += x[j] * x[j];
  }
  #pragma unroll
  for (int off = 16; off > 0; off >>= 1) ss += __shfl_xor(ss, off, 32);
  const float sc = rsqrtf(ss * (1.f / 256.f) + 1e-6f);
  #pragma unroll
  for (int j = 0; j < 8; j++)
    dt_out[(size_t)t * RR + lane + j * 32] =
        to_bf16(x[j] * sc * dtw[lane + j * 32]);

  // ---- B: 16 elems ----
  float xb = (lane < NN) ? row[RR + lane] : 0.f;
  float sb = xb * xb;
  #pragma unroll
  for (int off = 16; off > 0; off >>= 1) sb += __shfl_xor(sb, off, 32);
  const float scb = rsqrtf(sb * (1.f / 16.f) + 1e-6f);
  if (lane < NN) Bn[t * NN + lane] = xb * scb * bw[lane];

  // ---- C: 16 elems ----
  float xc = (lane < NN) ? row[RR + NN + lane] : 0.f;
  float scsum = xc * xc;
  #pragma unroll
  for (int off = 16; off > 0; off >>= 1) scsum += __shfl_xor(scsum, off, 32);
  const float scc = rsqrtf(scsum * (1.f / 16.f) + 1e-6f);
  if (lane < NN) Cn[t * NN + lane] = xc * scc * cwn[lane];
}

// ---------------------------------------------------------------------------
// Selective scan, one thread per channel d. 16-state recurrence in registers.
// Fuses D-skip and SiLU gating; emits bf16 A-matrix for out_proj.
// ---------------------------------------------------------------------------
__global__ __launch_bounds__(256)
void scan_kernel(const float* __restrict__ delta, const float* __restrict__ uf,
                 const float* __restrict__ Bn, const float* __restrict__ Cn,
                 const float* __restrict__ Ap, const float* __restrict__ Dp,
                 const float* __restrict__ proj, __bf16* __restrict__ yb) {
  const int d = blockIdx.x * 256 + threadIdx.x;
  float a[NN], st[NN];
  #pragma unroll
  for (int n = 0; n < NN; n++) {
    a[n] = Ap[d * NN + n];
    st[n] = 0.f;
  }
  const float dparam = Dp[d];
  for (int t = 0; t < LL; t++) {
    const float dl = delta[(size_t)t * DD + d];
    const float ut = uf[(size_t)t * DD + d];
    const float du = dl * ut;
    float y = 0.f;
    #pragma unroll
    for (int n = 0; n < NN; n++) {
      const float dA = __expf(dl * a[n]);
      st[n] = dA * st[n] + du * Bn[t * NN + n];
      y += st[n] * Cn[t * NN + n];
    }
    y += ut * dparam;
    const float g = proj[(size_t)t * (2 * DD) + DD + d];
    const float gated = y * (g / (1.f + __expf(-g)));
    yb[(size_t)t * DD + d] = to_bf16(gated);
  }
}

// ---------------------------------------------------------------------------
// Host-side orchestration
// ---------------------------------------------------------------------------
extern "C" void kernel_launch(void* const* d_in, const int* in_sizes, int n_in,
                              void* d_out, int out_size, void* d_ws, size_t ws_size,
                              hipStream_t stream) {
  const float* hs    = (const float*)d_in[0];   // [L, H]
  const float* Win   = (const float*)d_in[1];   // [2D, H]
  const float* convw = (const float*)d_in[2];   // [D, K]
  const float* convb = (const float*)d_in[3];   // [D]
  const float* Wx    = (const float*)d_in[4];   // [RN, D]
  const float* dtlnw = (const float*)d_in[5];   // [R]
  const float* blnw  = (const float*)d_in[6];   // [N]
  const float* clnw  = (const float*)d_in[7];   // [N]
  const float* Wdt   = (const float*)d_in[8];   // [D, R]
  const float* dtb   = (const float*)d_in[9];   // [D]
  const float* Ap    = (const float*)d_in[10];  // [D, N]
  const float* Dp    = (const float*)d_in[11];  // [D]
  const float* Wout  = (const float*)d_in[12];  // [H, D]
  float* out = (float*)d_out;                   // [L, H]

  char* ws = (char*)d_ws;
  size_t off = 0;
  auto alloc = [&](size_t bytes) -> void* {
    void* p = ws + off;
    off = (off + bytes + 255) & ~(size_t)255;
    return p;
  };

  __bf16* hs_b   = (__bf16*)alloc((size_t)LL * HH * 2);
  __bf16* Win_b  = (__bf16*)alloc((size_t)2 * DD * HH * 2);
  __bf16* Wx_b   = (__bf16*)alloc((size_t)RN * DD * 2);
  __bf16* Wdt_b  = (__bf16*)alloc((size_t)DD * RR * 2);
  __bf16* Wout_b = (__bf16*)alloc((size_t)HH * DD * 2);
  float*  projf  = (float*)alloc((size_t)LL * 2 * DD * 4);
  float*  uf     = (float*)alloc((size_t)LL * DD * 4);
  __bf16* ub     = (__bf16*)alloc((size_t)LL * DD * 2);
  float*  ssm    = (float*)alloc((size_t)LL * RN * 4);
  __bf16* dt_b   = (__bf16*)alloc((size_t)LL * RR * 2);
  float*  Bnrm   = (float*)alloc((size_t)LL * NN * 4);
  float*  Cnrm   = (float*)alloc((size_t)LL * NN * 4);
  float*  deltaf = (float*)alloc((size_t)LL * DD * 4);
  __bf16* yb     = (__bf16*)alloc((size_t)LL * DD * 2);

  auto cvt = [&](const float* s, __bf16* d, size_t n) {
    f32_to_bf16_kernel<<<(unsigned)(n / 1024), 256, 0, stream>>>(s, d, n);
  };
  cvt(hs, hs_b, (size_t)LL * HH);
  cvt(Win, Win_b, (size_t)2 * DD * HH);
  cvt(Wx, Wx_b, (size_t)RN * DD);
  cvt(Wdt, Wdt_b, (size_t)DD * RR);
  cvt(Wout, Wout_b, (size_t)HH * DD);

  // 1) in_proj: proj[L, 2D] = hs @ Win^T
  {
    dim3 grid((2 * DD + BN - 1) / BN, (LL + BM - 1) / BM);
    gemm_bf16_wmma<0><<<grid, 256, 0, stream>>>(hs_b, Win_b, nullptr, projf,
                                                LL, 2 * DD, HH);
  }
  // 2) causal conv + SiLU -> u
  conv_silu_kernel<<<(LL * DD) / 256, 256, 0, stream>>>(projf, convw, convb,
                                                        uf, ub);
  // 3) x_proj: ssm[L, RN] = u @ Wx^T
  {
    dim3 grid((RN + BN - 1) / BN, (LL + BM - 1) / BM);
    gemm_bf16_wmma<0><<<grid, 256, 0, stream>>>(ub, Wx_b, nullptr, ssm,
                                                LL, RN, DD);
  }
  // 4) RMSNorms on dt / B / C
  rmsnorm_kernel<<<LL / 8, 256, 0, stream>>>(ssm, dtlnw, blnw, clnw,
                                             dt_b, Bnrm, Cnrm);
  // 5) dt_proj + softplus: delta[L, D] = softplus(dt @ Wdt^T + dt_bias)
  {
    dim3 grid((DD + BN - 1) / BN, (LL + BM - 1) / BM);
    gemm_bf16_wmma<1><<<grid, 256, 0, stream>>>(dt_b, Wdt_b, dtb, deltaf,
                                                LL, DD, RR);
  }
  // 6) selective scan + D-skip + SiLU gate -> y (bf16)
  scan_kernel<<<DD / 256, 256, 0, stream>>>(deltaf, uf, Bnrm, Cnrm, Ap, Dp,
                                            projf, yb);
  // 7) out_proj: out[L, H] = y @ Wout^T
  {
    dim3 grid((HH + BN - 1) / BN, (LL + BM - 1) / BM);
    gemm_bf16_wmma<0><<<grid, 256, 0, stream>>>(yb, Wout_b, nullptr, out,
                                                LL, HH, DD);
  }
}
